// MotifEnergy_87909390615161
// MI455X (gfx1250) — matrix operand order repaired and checked by reading, not compiled, not run on hardware
//
#include <hip/hip_runtime.h>
#include <hip/hip_bf16.h>
#include <math.h>

typedef __attribute__((ext_vector_type(16))) _Float16 v16h;
typedef __attribute__((ext_vector_type(8)))  float    v8f;

#define R_DIM 4
#define D_DIM 16
#define BETA_MAX_F 5.0f

__device__ __forceinline__ float softplus_f(float x) {
  return log1pf(expf(x));   // inputs are uniform(0,1): no overflow concern
}

// Order-preserving map float -> uint32 so segment-max becomes a single
// branchless global_atomic_max_u32 (no divergent two-sided atomic).
__device__ __forceinline__ unsigned fkey_encode(float f) {
  int b = __float_as_int(f);
  return (b >= 0) ? ((unsigned)b ^ 0x80000000u) : ~(unsigned)b;
}
__device__ __forceinline__ float fkey_decode(unsigned k) {
  int b = (k & 0x80000000u) ? (int)(k ^ 0x80000000u) : (int)~k;
  return __int_as_float(b);
}
#define FKEY_NEG_INF 0x007FFFFFu   // fkey_encode(-inf)

// -------- init: seg_max = key(-inf), sums = 0, out = 0 --------
__global__ void init_kernel(unsigned* __restrict__ segmax, float* __restrict__ sums,
                            float* __restrict__ out, int nr, int gr) {
  int i = blockIdx.x * blockDim.x + threadIdx.x;
  if (i < nr) { segmax[i] = FKEY_NEG_INF; sums[i] = 0.0f; }
  if (i < gr) out[i] = 0.0f;
}

// -------- pass A: WMMA motif energies + atomic segment max --------
// One wave = one tile of 16 triangles. For each r: D = A(16x32,f16) x B(32x16,f16),
// diag(D)[m] = ell[m,r]*8 for the 16 triangles of the tile.
__global__ __launch_bounds__(256) void
motif_wmma_kernel(const int* __restrict__ c3, const int* __restrict__ u3,
                  const int* __restrict__ v3, const int* __restrict__ ttau,
                  const float* __restrict__ Q3, const float* __restrict__ K3,
                  const float* __restrict__ Tp, const float* __restrict__ raw_beta,
                  float* __restrict__ zbuf, unsigned* __restrict__ segmax, int M) {
  const int lane = threadIdx.x & 31;
  const int wave = threadIdx.x >> 5;
  const int tile = blockIdx.x * (blockDim.x >> 5) + wave;
  const int numTiles = (M + 15) >> 4;
  if (tile >= numTiles) return;

  const float beta = fminf(softplus_f(raw_beta[0]), BETA_MAX_F);
  const float betaOverScale = beta * 0.125f;   // beta / sqrt(R*d) = beta/8

  // lanes 0..15 -> rows 0..15 (K 0..7); lanes 16..31 -> rows 0..15 (K 8..15)
  const int rowId = (lane < 16) ? lane : (lane - 16);
  const int tl = tile * 16 + rowId;            // this lane's triangle
  const int m  = (tl < M) ? tl : (M - 1);      // clamp tail (writes masked later)

  const int c  = c3[m];
  const int u  = u3[m];
  const int v  = v3[m];
  const int tt = ttau[m];

  // B matrix (32x16 f16): lane l holds column n=l%16, elements 0..7 = T_sel[n][8*(l/16)..+7]; elements 8..15 zero.
  v16h b = {};
  {
    const float4* tp = (const float4*)(Tp + (size_t)tt * D_DIM + ((lane < 16) ? 0 : 8));
    float4 t0 = tp[0], t1 = tp[1];
    b[0]  = (_Float16)t0.x; b[1]  = (_Float16)t0.y; b[2]  = (_Float16)t0.z; b[3]  = (_Float16)t0.w;
    b[4]  = (_Float16)t1.x; b[5]  = (_Float16)t1.y; b[6]  = (_Float16)t1.z; b[7]  = (_Float16)t1.w;
  }

  const int dOff = (lane < 16) ? 0 : 8;        // which half of d this lane supplies
  const float* qbase = Q3 + ((size_t)c * R_DIM) * D_DIM + dOff;
  const float* ubase = K3 + ((size_t)u * R_DIM) * D_DIM + dOff;
  const float* vbase = K3 + ((size_t)v * R_DIM) * D_DIM + dOff;

  // diagonal D[m,m]: lanes 0..7 -> acc[lane]; lanes 24..31 -> acc[lane-24]
  const int diagIdx = (lane < 8) ? lane : ((lane >= 24) ? (lane - 24) : -1);

  #pragma unroll
  for (int r = 0; r < R_DIM; ++r) {
    const float4* qp = (const float4*)(qbase + r * D_DIM);
    const float4* up = (const float4*)(ubase + r * D_DIM);
    const float4* vp = (const float4*)(vbase + r * D_DIM);
    float4 q0 = qp[0],  q1 = qp[1];
    float4 a0 = up[0],  a1 = up[1];
    float4 s0 = vp[0],  s1 = vp[1];

    v16h a = {};
    a[0] = (_Float16)(q0.x * a0.x * s0.x);
    a[1] = (_Float16)(q0.y * a0.y * s0.y);
    a[2] = (_Float16)(q0.z * a0.z * s0.z);
    a[3] = (_Float16)(q0.w * a0.w * s0.w);
    a[4] = (_Float16)(q1.x * a1.x * s1.x);
    a[5] = (_Float16)(q1.y * a1.y * s1.y);
    a[6] = (_Float16)(q1.z * a1.z * s1.z);
    a[7] = (_Float16)(q1.w * a1.w * s1.w);

    v8f acc = {};
    acc = __builtin_amdgcn_wmma_f32_16x16x32_f16(
        /*neg_a=*/false, a, /*neg_b=*/false, b,
        /*c_mod=*/(short)0, acc, /*reuse_a=*/false, /*reuse_b=*/false);

    float zv = 0.0f;
    #pragma unroll
    for (int j = 0; j < 8; ++j) zv = (diagIdx == j) ? acc[j] : zv;

    float z = betaOverScale * zv;
    if (diagIdx >= 0 && tl < M) {
      zbuf[(size_t)tl * R_DIM + r] = z;
      atomicMax(&segmax[c * R_DIM + r], fkey_encode(z));  // single branchless atomic
    }
  }
}

// -------- pass B: sums[c,r] += exp(z - seg_max[c,r]) --------
__global__ void expsum_kernel(const int* __restrict__ c3, const float* __restrict__ zbuf,
                              const unsigned* __restrict__ segmax, float* __restrict__ sums,
                              int MR) {
  int i = blockIdx.x * blockDim.x + threadIdx.x;
  if (i >= MR) return;
  int m = i >> 2, r = i & 3;
  float z = zbuf[i];
  int c = c3[m];
  float sm = fkey_decode(segmax[c * R_DIM + r]);  // segment non-empty -> finite
  atomicAdd(&sums[c * R_DIM + r], expf(z - sm));
}

// -------- pass C: lse per (node,r), scatter into graphs --------
__global__ void lse_scatter_kernel(const int* __restrict__ batch,
                                   const unsigned* __restrict__ segmax,
                                   const float* __restrict__ sums,
                                   const float* __restrict__ raw_lambda,
                                   const float* __restrict__ raw_beta,
                                   float* __restrict__ out, int NR) {
  int i = blockIdx.x * blockDim.x + threadIdx.x;
  if (i >= NR) return;
  int n = i >> 2, r = i & 3;
  float sm = fkey_decode(segmax[i]);
  float s  = sums[i];
  float safe = isfinite(sm) ? sm : 0.0f;
  float lse  = (s > 0.0f) ? (safe + logf(s)) : 0.0f;
  if (lse != 0.0f) {
    float lam  = softplus_f(raw_lambda[0]);
    float beta = fminf(softplus_f(raw_beta[0]), BETA_MAX_F);
    atomicAdd(&out[batch[n] * R_DIM + r], (lam / beta) * lse);
  }
}

extern "C" void kernel_launch(void* const* d_in, const int* in_sizes, int n_in,
                              void* d_out, int out_size, void* d_ws, size_t ws_size,
                              hipStream_t stream) {
  const int*   c3    = (const int*)d_in[0];
  const int*   u3    = (const int*)d_in[1];
  const int*   v3    = (const int*)d_in[2];
  const int*   ttau  = (const int*)d_in[3];
  const int*   batch = (const int*)d_in[4];
  const float* Q3    = (const float*)d_in[5];
  const float* K3    = (const float*)d_in[6];
  const float* Tp    = (const float*)d_in[7];
  const float* rlam  = (const float*)d_in[8];
  const float* rbeta = (const float*)d_in[9];

  const int M  = in_sizes[0];       // #triangles
  const int N  = in_sizes[4];       // #nodes (batch has N entries)
  const int NR = N * R_DIM;
  const int MR = M * R_DIM;
  const int GR = out_size;          // num_graphs * R
  float* out = (float*)d_out;

  // workspace partition: [segmax NR u32][sums NR f32][zbuf MR f32] (~19.2 MB)
  unsigned* segmax = (unsigned*)d_ws;
  float*    sums   = (float*)(segmax + NR);
  float*    zbuf   = sums + NR;
  (void)ws_size; (void)n_in;

  const int initN = (NR > GR) ? NR : GR;
  init_kernel<<<(initN + 255) / 256, 256, 0, stream>>>(segmax, sums, out, NR, GR);

  const int numTiles = (M + 15) / 16;
  const int wavesPerBlock = 8;      // 256 threads = 8 wave32 tiles
  const int blocksA = (numTiles + wavesPerBlock - 1) / wavesPerBlock;
  motif_wmma_kernel<<<blocksA, 256, 0, stream>>>(c3, u3, v3, ttau, Q3, K3, Tp,
                                                 rbeta, zbuf, segmax, M);

  expsum_kernel<<<(MR + 255) / 256, 256, 0, stream>>>(c3, zbuf, segmax, sums, MR);

  lse_scatter_kernel<<<(NR + 255) / 256, 256, 0, stream>>>(batch, segmax, sums,
                                                           rlam, rbeta, out, NR);
}
